// Top_K_MoE_9904194584778
// MI455X (gfx1250) — compile-verified, matrix operand order checked by reading, not run on hardware
//
#include <hip/hip_runtime.h>

// ---------------------------------------------------------------- constants
#define T_TOKENS   8192
#define D_IN       1024
#define D_INTER    4096
#define D_OUT      4096
#define NEXP       8
#define NPAIRS     (T_TOKENS * 2)   // token-expert pairs (top-2)
#define MAXP_PER_E 8192             // max pairs per expert

#define BM  128
#define BN  128
#define BK  32
#define LDP 40                      // LDS pitch (elements) for BK=32 tiles

typedef __attribute__((ext_vector_type(16))) __bf16 v16bf;
typedef __attribute__((ext_vector_type(8)))  __bf16 v8bf;
typedef __attribute__((ext_vector_type(2)))  __bf16 v2bf;
typedef __attribute__((ext_vector_type(8)))  float  v8f;

// workspace layout (bytes)
static constexpr size_t OFF_CNT  = 0;
static constexpr size_t OFF_SEL  = 256;
static constexpr size_t OFF_W    = OFF_SEL  + (size_t)NPAIRS * 4;
static constexpr size_t OFF_LIST = OFF_W    + (size_t)NPAIRS * 4;
static constexpr size_t OFF_H    = OFF_LIST + (size_t)NEXP * MAXP_PER_E * 4;
static constexpr size_t OFF_Y    = OFF_H    + (size_t)NPAIRS * D_INTER * 2;

// ---------------------------------------------------------------- init
__global__ void moe_init(int* __restrict__ cnt) {
    if (threadIdx.x < NEXP) cnt[threadIdx.x] = 0;
}

// ---------------------------------------------------------------- router
// one wave (32 lanes) per token; lanes split the D=1024 dot products
__global__ void __launch_bounds__(256)
moe_router(const float* __restrict__ x, const float* __restrict__ gw,
           float* __restrict__ logits_out, int* __restrict__ sel_pair,
           float* __restrict__ w_pair, int* __restrict__ lists,
           int* __restrict__ cnt)
{
    const int wid  = threadIdx.x >> 5;
    const int lane = threadIdx.x & 31;
    const int t    = blockIdx.x * 8 + wid;

    float acc[NEXP];
#pragma unroll
    for (int e = 0; e < NEXP; ++e) acc[e] = 0.f;

    const float* xr = x + (size_t)t * D_IN;
    for (int d = lane; d < D_IN; d += 32) {
        const float xv = xr[d];
        const float* g = gw + (size_t)d * NEXP;
#pragma unroll
        for (int e = 0; e < NEXP; ++e) acc[e] = fmaf(xv, g[e], acc[e]);
    }
#pragma unroll
    for (int e = 0; e < NEXP; ++e) {
#pragma unroll
        for (int off = 16; off > 0; off >>= 1)
            acc[e] += __shfl_xor(acc[e], off, 32);
    }

    if (lane == 0) {
#pragma unroll
        for (int e = 0; e < NEXP; ++e) logits_out[(size_t)t * NEXP + e] = acc[e];
        // top-1 (ties -> lowest index, matches jax.lax.top_k)
        int   i1 = 0; float l1 = acc[0];
#pragma unroll
        for (int e = 1; e < NEXP; ++e) if (acc[e] > l1) { l1 = acc[e]; i1 = e; }
        int   i2 = -1; float l2 = -3.4e38f;
#pragma unroll
        for (int e = 0; e < NEXP; ++e)
            if (e != i1 && acc[e] > l2) { l2 = acc[e]; i2 = e; }
        // normalized top-2 softmax weights: p1/(p1+p2) = 1/(1+exp(l2-l1))
        const float w1 = 1.f / (1.f + __expf(l2 - l1));
        const float w2 = 1.f - w1;

        const int p0 = 2 * t, p1 = 2 * t + 1;
        sel_pair[p0] = i1; w_pair[p0] = w1;
        sel_pair[p1] = i2; w_pair[p1] = w2;
        int pos = atomicAdd(&cnt[i1], 1);
        lists[i1 * MAXP_PER_E + pos] = p0;
        pos = atomicAdd(&cnt[i2], 1);
        lists[i2 * MAXP_PER_E + pos] = p1;
    }
}

// ---------------------------------------------------------------- helpers
__device__ __forceinline__ v16bf frag_from_lds(const __bf16* base, int koff) {
    const v8bf lo = *(const v8bf*)(base + koff);
    const v8bf hi = *(const v8bf*)(base + koff + 16);
    return __builtin_shufflevector(lo, hi, 0, 1, 2, 3, 4, 5, 6, 7,
                                           8, 9, 10, 11, 12, 13, 14, 15);
}

// pack two f32 (consecutive k of one column) -> one b32 LDS store
__device__ __forceinline__ void store_kpair(__bf16* dst, float lo, float hi) {
    v2bf p;
    p.x = (__bf16)lo;
    p.y = (__bf16)hi;
    *(v2bf*)dst = p;
}

// store a B k-pair micro-tile: bR[0..1] = row k cols c0..c0+7, bR[2..3] = row k+1
__device__ __forceinline__ void stage_B_pairs(__bf16 (*Bsb)[LDP], int bc0, int bk0,
                                              const float4* bR) {
    store_kpair(&Bsb[bc0 + 0][bk0], bR[0].x, bR[2].x);
    store_kpair(&Bsb[bc0 + 1][bk0], bR[0].y, bR[2].y);
    store_kpair(&Bsb[bc0 + 2][bk0], bR[0].z, bR[2].z);
    store_kpair(&Bsb[bc0 + 3][bk0], bR[0].w, bR[2].w);
    store_kpair(&Bsb[bc0 + 4][bk0], bR[1].x, bR[3].x);
    store_kpair(&Bsb[bc0 + 5][bk0], bR[1].y, bR[3].y);
    store_kpair(&Bsb[bc0 + 6][bk0], bR[1].z, bR[3].z);
    store_kpair(&Bsb[bc0 + 7][bk0], bR[1].w, bR[3].w);
}

__device__ __forceinline__ void stage_A_f32(__bf16* dst, const float4* aR) {
#pragma unroll
    for (int j = 0; j < 4; ++j) {
        dst[4 * j + 0] = (__bf16)aR[j].x;
        dst[4 * j + 1] = (__bf16)aR[j].y;
        dst[4 * j + 2] = (__bf16)aR[j].z;
        dst[4 * j + 3] = (__bf16)aR[j].w;
    }
}

// ---------------------------------------------------------------- GEMM1
// h[p, :] = relu(x[token(p), :] @ W1[e] + b1[e])   (bf16 out)
// 128x128 tile / block, 8 waves of 32x64, ping-pong LDS, 8 wmma / K-step
__global__ void __launch_bounds__(256)
moe_gemm1(const float* __restrict__ x, const float* __restrict__ W1,
          const float* __restrict__ b1, const int* __restrict__ lists,
          const int* __restrict__ cnt, __bf16* __restrict__ h)
{
    __shared__ int    rows[BM];
    __shared__ __bf16 As[2][BM][LDP];
    __shared__ __bf16 Bs[2][BN][LDP];

    const int e     = blockIdx.z;
    const int mbase = blockIdx.y * BM;
    const int ne    = cnt[e];
    if (mbase >= ne) return;
    const int nbase = blockIdx.x * BN;
    const int tid   = threadIdx.x;

    if (tid < BM) {
        const int idx = mbase + tid;
        rows[tid] = (idx < ne) ? lists[e * MAXP_PER_E + idx] : -1;
    }
    __syncthreads();

    const int lane  = tid & 31;
    const int wid   = tid >> 5;
    const int waveM = (wid & 3) * 32;
    const int waveN = (wid >> 2) * 64;
    const int lrow  = lane & 15;
    const int koff  = (lane < 16) ? 0 : 8;

    // A staging map: 2 threads per row, 16 contiguous k each
    const int arow = tid >> 1;
    const int ak0  = (tid & 1) * 16;
    const int ap   = rows[arow];
    const float* axr = (ap >= 0) ? (x + (size_t)(ap >> 1) * D_IN) : x;

    // B staging map: each thread stages a k-pair x 8 columns
    const int bk0 = (tid >> 4) * 2;   // even k in [0, 30]
    const int bc0 = (tid & 15) * 8;   // 8 consecutive columns

    const float* W1e = W1 + (size_t)e * D_IN * D_INTER;

    // zero invalid (padding) A rows once for both buffers
    if (ap < 0) {
#pragma unroll
        for (int j = 0; j < 16; ++j) {
            As[0][arow][ak0 + j] = (__bf16)0.f;
            As[1][arow][ak0 + j] = (__bf16)0.f;
        }
    }

    v8f acc[2][4] = {};

    // ---- prologue: stage K-tile 0 into buffer 0
    {
        if (ap >= 0) {
            float4 aR[4];
            const float4* s = (const float4*)(axr + ak0);
#pragma unroll
            for (int j = 0; j < 4; ++j) aR[j] = s[j];
            stage_A_f32(&As[0][arow][ak0], aR);
        }
        float4 bR[4];
        const float4* s0 = (const float4*)(W1e + (size_t)bk0 * D_INTER + nbase + bc0);
        const float4* s1 = (const float4*)(W1e + (size_t)(bk0 + 1) * D_INTER + nbase + bc0);
        bR[0] = s0[0]; bR[1] = s0[1]; bR[2] = s1[0]; bR[3] = s1[1];
        stage_B_pairs(Bs[0], bc0, bk0, bR);
    }

    const int KT = D_IN / BK;
    for (int kt = 0; kt < KT; ++kt) {
        const int cur = kt & 1;
        const int nxt = cur ^ 1;
        __syncthreads();

        // issue next tile's global loads (overlap with wmma below)
        const bool more = (kt + 1 < KT);
        float4 aR[4], bR[4];
        if (more) {
            const int kk = (kt + 1) * BK;
            if (ap >= 0) {
                const float4* s = (const float4*)(axr + kk + ak0);
#pragma unroll
                for (int j = 0; j < 4; ++j) aR[j] = s[j];
            }
            const float4* s0 =
                (const float4*)(W1e + (size_t)(kk + bk0) * D_INTER + nbase + bc0);
            const float4* s1 =
                (const float4*)(W1e + (size_t)(kk + bk0 + 1) * D_INTER + nbase + bc0);
            bR[0] = s0[0]; bR[1] = s0[1]; bR[2] = s1[0]; bR[3] = s1[1];
        }

        // compute on current buffer
        v16bf afr[2], bfr[4];
#pragma unroll
        for (int mi = 0; mi < 2; ++mi)
            afr[mi] = frag_from_lds(&As[cur][waveM + mi * 16 + lrow][0], koff);
#pragma unroll
        for (int ni = 0; ni < 4; ++ni)
            bfr[ni] = frag_from_lds(&Bs[cur][waveN + ni * 16 + lrow][0], koff);
#pragma unroll
        for (int mi = 0; mi < 2; ++mi)
#pragma unroll
            for (int ni = 0; ni < 4; ++ni)
                acc[mi][ni] = __builtin_amdgcn_wmma_f32_16x16x32_bf16(
                    false, afr[mi], false, bfr[ni], (short)0, acc[mi][ni],
                    false, false);

        // convert + store next tile into the other buffer
        if (more) {
            if (ap >= 0) stage_A_f32(&As[nxt][arow][ak0], aR);
            stage_B_pairs(Bs[nxt], bc0, bk0, bR);
        }
    }

    // ---- epilogue: +bias, relu, bf16 scatter to h
    const int rofs = (lane < 16) ? 0 : 8;
#pragma unroll
    for (int ni = 0; ni < 4; ++ni) {
        const int   col  = nbase + waveN + ni * 16 + lrow;
        const float bias = b1[(size_t)e * D_INTER + col];
#pragma unroll
        for (int mi = 0; mi < 2; ++mi) {
#pragma unroll
            for (int j = 0; j < 8; ++j) {
                const int rl = waveM + mi * 16 + rofs + j;
                const int p  = rows[rl];
                if (p >= 0) {
                    float v = acc[mi][ni][j] + bias;
                    v = v > 0.f ? v : 0.f;
                    h[(size_t)p * D_INTER + col] = (__bf16)v;
                }
            }
        }
    }
}

// ---------------------------------------------------------------- GEMM2
// y[p, :] = h[p, :] @ W2[e] + b2[e]   (f32 out, pre-LN)
__global__ void __launch_bounds__(256)
moe_gemm2(const __bf16* __restrict__ h, const float* __restrict__ W2,
          const float* __restrict__ b2, const int* __restrict__ lists,
          const int* __restrict__ cnt, float* __restrict__ y)
{
    __shared__ int    rows[BM];
    __shared__ __bf16 As[2][BM][LDP];
    __shared__ __bf16 Bs[2][BN][LDP];

    const int e     = blockIdx.z;
    const int mbase = blockIdx.y * BM;
    const int ne    = cnt[e];
    if (mbase >= ne) return;
    const int nbase = blockIdx.x * BN;
    const int tid   = threadIdx.x;

    if (tid < BM) {
        const int idx = mbase + tid;
        rows[tid] = (idx < ne) ? lists[e * MAXP_PER_E + idx] : -1;
    }
    __syncthreads();

    const int lane  = tid & 31;
    const int wid   = tid >> 5;
    const int waveM = (wid & 3) * 32;
    const int waveN = (wid >> 2) * 64;
    const int lrow  = lane & 15;
    const int koff  = (lane < 16) ? 0 : 8;

    const int arow = tid >> 1;
    const int ak0  = (tid & 1) * 16;
    const int ap   = rows[arow];
    const __bf16* ahr = (ap >= 0) ? (h + (size_t)ap * D_INTER) : h;

    const int bk0 = (tid >> 4) * 2;
    const int bc0 = (tid & 15) * 8;

    const float* W2e = W2 + (size_t)e * D_INTER * D_OUT;

    if (ap < 0) {
#pragma unroll
        for (int j = 0; j < 16; ++j) {
            As[0][arow][ak0 + j] = (__bf16)0.f;
            As[1][arow][ak0 + j] = (__bf16)0.f;
        }
    }

    v8f acc[2][4] = {};

    // ---- prologue: stage K-tile 0 into buffer 0
    {
        if (ap >= 0) {
            const v8bf* s = (const v8bf*)(ahr + ak0);
            *(v8bf*)&As[0][arow][ak0]     = s[0];
            *(v8bf*)&As[0][arow][ak0 + 8] = s[1];
        }
        float4 bR[4];
        const float4* s0 = (const float4*)(W2e + (size_t)bk0 * D_OUT + nbase + bc0);
        const float4* s1 = (const float4*)(W2e + (size_t)(bk0 + 1) * D_OUT + nbase + bc0);
        bR[0] = s0[0]; bR[1] = s0[1]; bR[2] = s1[0]; bR[3] = s1[1];
        stage_B_pairs(Bs[0], bc0, bk0, bR);
    }

    const int KT = D_INTER / BK;
    for (int kt = 0; kt < KT; ++kt) {
        const int cur = kt & 1;
        const int nxt = cur ^ 1;
        __syncthreads();

        const bool more = (kt + 1 < KT);
        v8bf  aRh[2];
        float4 bR[4];
        if (more) {
            const int kk = (kt + 1) * BK;
            if (ap >= 0) {
                const v8bf* s = (const v8bf*)(ahr + kk + ak0);
                aRh[0] = s[0]; aRh[1] = s[1];
            }
            const float4* s0 =
                (const float4*)(W2e + (size_t)(kk + bk0) * D_OUT + nbase + bc0);
            const float4* s1 =
                (const float4*)(W2e + (size_t)(kk + bk0 + 1) * D_OUT + nbase + bc0);
            bR[0] = s0[0]; bR[1] = s0[1]; bR[2] = s1[0]; bR[3] = s1[1];
        }

        v16bf afr[2], bfr[4];
#pragma unroll
        for (int mi = 0; mi < 2; ++mi)
            afr[mi] = frag_from_lds(&As[cur][waveM + mi * 16 + lrow][0], koff);
#pragma unroll
        for (int ni = 0; ni < 4; ++ni)
            bfr[ni] = frag_from_lds(&Bs[cur][waveN + ni * 16 + lrow][0], koff);
#pragma unroll
        for (int mi = 0; mi < 2; ++mi)
#pragma unroll
            for (int ni = 0; ni < 4; ++ni)
                acc[mi][ni] = __builtin_amdgcn_wmma_f32_16x16x32_bf16(
                    false, afr[mi], false, bfr[ni], (short)0, acc[mi][ni],
                    false, false);

        if (more) {
            if (ap >= 0) {
                *(v8bf*)&As[nxt][arow][ak0]     = aRh[0];
                *(v8bf*)&As[nxt][arow][ak0 + 8] = aRh[1];
            }
            stage_B_pairs(Bs[nxt], bc0, bk0, bR);
        }
    }

    const int rofs = (lane < 16) ? 0 : 8;
#pragma unroll
    for (int ni = 0; ni < 4; ++ni) {
        const int   col  = nbase + waveN + ni * 16 + lrow;
        const float bias = b2[(size_t)e * D_OUT + col];
#pragma unroll
        for (int mi = 0; mi < 2; ++mi) {
#pragma unroll
            for (int j = 0; j < 8; ++j) {
                const int rl = waveM + mi * 16 + rofs + j;
                const int p  = rows[rl];
                if (p >= 0)
                    y[(size_t)p * D_OUT + col] = acc[mi][ni][j] + bias;
            }
        }
    }
}

// ---------------------------------------------------------------- LN + combine
// out[t] = sum_k w[2t+k] * LayerNorm(y[2t+k]) — one block per token.
__global__ void __launch_bounds__(256)
moe_ln(const float* __restrict__ y, const int* __restrict__ sel_pair,
       const float* __restrict__ w_pair, const float* __restrict__ ln_g,
       const float* __restrict__ ln_b, float* __restrict__ out)
{
    __shared__ float red[256];
    const int t   = blockIdx.x;
    const int tid = threadIdx.x;

    float accv[16];
#pragma unroll
    for (int i = 0; i < 16; ++i) accv[i] = 0.f;

    for (int k = 0; k < 2; ++k) {
        const int   p  = 2 * t + k;
        const int   e  = sel_pair[p];
        const float w  = w_pair[p];
        const float* yr = y + (size_t)p * D_OUT;

        float s = 0.f, s2 = 0.f;
#pragma unroll
        for (int i = 0; i < 16; ++i) {
            const float v = yr[tid + i * 256];
            s += v; s2 = fmaf(v, v, s2);
        }
        red[tid] = s; __syncthreads();
        for (int off = 128; off > 0; off >>= 1) {
            if (tid < off) red[tid] += red[tid + off];
            __syncthreads();
        }
        const float mu = red[0] * (1.f / D_OUT);
        __syncthreads();
        red[tid] = s2; __syncthreads();
        for (int off = 128; off > 0; off >>= 1) {
            if (tid < off) red[tid] += red[tid + off];
            __syncthreads();
        }
        const float var = red[0] * (1.f / D_OUT) - mu * mu;
        __syncthreads();
        const float rs = rsqrtf(var + 1e-5f);

#pragma unroll
        for (int i = 0; i < 16; ++i) {
            const int   c = tid + i * 256;
            const float v =
                (yr[c] - mu) * rs * ln_g[(size_t)e * D_OUT + c] +
                ln_b[(size_t)e * D_OUT + c];
            accv[i] = fmaf(w, v, accv[i]);
        }
    }
#pragma unroll
    for (int i = 0; i < 16; ++i)
        out[(size_t)t * D_OUT + tid + i * 256] = accv[i];
}

// ---------------------------------------------------------------- launch
extern "C" void kernel_launch(void* const* d_in, const int* in_sizes, int n_in,
                              void* d_out, int out_size, void* d_ws, size_t ws_size,
                              hipStream_t stream)
{
    (void)in_sizes; (void)n_in; (void)out_size; (void)ws_size;

    const float* x      = (const float*)d_in[0];  // [4,2048,1024]
    const float* gate_w = (const float*)d_in[1];  // [1024,8]
    const float* W1     = (const float*)d_in[2];  // [8,1024,4096]
    const float* b1     = (const float*)d_in[3];  // [8,4096]
    const float* W2     = (const float*)d_in[4];  // [8,4096,4096]
    const float* b2     = (const float*)d_in[5];  // [8,4096]
    const float* ln_g   = (const float*)d_in[6];  // [8,4096]
    const float* ln_b   = (const float*)d_in[7];  // [8,4096]

    float* out    = (float*)d_out;                         // [8192,4096]
    float* logits = out + (size_t)T_TOKENS * D_OUT;        // [8192,8]

    char*   ws       = (char*)d_ws;
    int*    cnt      = (int*)(ws + OFF_CNT);
    int*    sel_pair = (int*)(ws + OFF_SEL);
    float*  w_pair   = (float*)(ws + OFF_W);
    int*    lists    = (int*)(ws + OFF_LIST);
    __bf16* hbuf     = (__bf16*)(ws + OFF_H);
    float*  ybuf     = (float*)(ws + OFF_Y);

    moe_init<<<1, 32, 0, stream>>>(cnt);
    moe_router<<<T_TOKENS / 8, 256, 0, stream>>>(x, gate_w, logits, sel_pair,
                                                 w_pair, lists, cnt);
    {
        dim3 grid(D_INTER / BN, MAXP_PER_E / BM, NEXP);  // (32, 64, 8)
        moe_gemm1<<<grid, 256, 0, stream>>>(x, W1, b1, lists, cnt, hbuf);
    }
    {
        dim3 grid(D_OUT / BN, MAXP_PER_E / BM, NEXP);    // (32, 64, 8)
        moe_gemm2<<<grid, 256, 0, stream>>>(hbuf, W2, b2, lists, cnt, ybuf);
    }
    moe_ln<<<T_TOKENS, 256, 0, stream>>>(ybuf, sel_pair, w_pair, ln_g, ln_b, out);
}
